// NetA_NodeOnly_16355235463255
// MI455X (gfx1250) — compile-verified, matrix operand order checked by reading, not run on hardware
//
#include <hip/hip_runtime.h>
#include <hip/hip_bf16.h>

// ---------------------------------------------------------------------------
// Types for CDNA5 WMMA (wave32, 16x16x32 bf16 -> f32)
// ---------------------------------------------------------------------------
typedef __attribute__((ext_vector_type(16))) __bf16 bf16x16;
typedef __attribute__((ext_vector_type(8)))  __bf16 bf16x8;
typedef __attribute__((ext_vector_type(8)))  float  f32x8;

// f32 -> bf16 bits, round-to-nearest-even
__device__ inline unsigned bf_bits(float f) {
  unsigned u = __float_as_uint(f);
  return (u + 0x7FFFu + ((u >> 16) & 1u)) >> 16;
}
// pack two f32 into one dword of two bf16 (lo = first)
__device__ inline unsigned pk_bf16x2(float a, float b) {
  return (bf_bits(a) & 0xFFFFu) | (bf_bits(b) << 16);
}

// order-preserving encoding for float atomic max over unsigned ints
__device__ inline unsigned f2ord(float f) {
  unsigned u = __float_as_uint(f);
  return (u & 0x80000000u) ? ~u : (u | 0x80000000u);
}
__device__ inline float ord2f(unsigned u) {
  return (u & 0x80000000u) ? __uint_as_float(u & 0x7FFFFFFFu) : __uint_as_float(~u);
}

#define TM 128
#define TN 64
#define TK 32
#define KPAD 8   // bf16 elements of row padding (80B row stride)

// ---------------------------------------------------------------------------
// Tiled WMMA GEMM:  C[M,Nout] = act(A[M,K] @ W[K,Nout] + bias)
// block = 256 threads = 8 waves; wave grid 2(M) x 4(N); each wave: 4 WMMAs.
// LDS tiles are staged as packed bf16; fragments load as 2x ds_load_b128.
// ---------------------------------------------------------------------------
__global__ __launch_bounds__(256)
void gemm_bias_kernel(const float* __restrict__ A, const float* __restrict__ W,
                      const float* __restrict__ bias, float* __restrict__ C,
                      int M, int K, int Nout, int do_relu) {
  __shared__ __attribute__((aligned(16))) __bf16 As[TM][TK + KPAD];  // [M][K]
  __shared__ __attribute__((aligned(16))) __bf16 Bs[TN][TK + KPAD];  // [N][K] (transposed)

  const int tid  = threadIdx.x;
  const int wid  = tid >> 5;
  const int lane = tid & 31;
  const int wm   = wid >> 2;   // 0..1
  const int wn   = wid & 3;    // 0..3
  const int m0   = blockIdx.y * TM;
  const int n0   = blockIdx.x * TN;

  union Acc { f32x8 v; float f[8]; };
  Acc acc[4];
  #pragma unroll
  for (int s = 0; s < 4; ++s)
    #pragma unroll
    for (int i = 0; i < 8; ++i) acc[s].f[i] = 0.f;

  for (int k0 = 0; k0 < K; k0 += TK) {
    // ---- stage A tile: 128x32 f32 -> packed bf16 pairs (2048 dword stores)
    for (int li = tid; li < (TM * TK) / 2; li += 256) {
      int r = li >> 4;            // 0..127
      int c = (li & 15) << 1;     // 0,2,..,30
      int gm = m0 + r, gk = k0 + c;
      float f0 = 0.f, f1 = 0.f;
      if (gm < M) {
        const float* ap = &A[(long)gm * K + gk];
        if (gk     < K) f0 = ap[0];
        if (gk + 1 < K) f1 = ap[1];
      }
      *(unsigned*)(&As[r][c]) = pk_bf16x2(f0, f1);
    }
    // ---- stage B tile transposed: W[K][Nout] slab -> Bs[N][K] (1024 stores)
    for (int li = tid; li < (TN * TK) / 2; li += 256) {
      int n  = li & 63;           // 0..63 (fastest across threads: coalesced)
      int kk = (li >> 6) << 1;    // 0,2,..,30
      int gn = n0 + n, gk = k0 + kk;
      float f0 = 0.f, f1 = 0.f;
      if (gn < Nout) {
        if (gk     < K) f0 = W[(long)gk * Nout + gn];
        if (gk + 1 < K) f1 = W[(long)(gk + 1) * Nout + gn];
      }
      *(unsigned*)(&Bs[n][kk]) = pk_bf16x2(f0, f1);
    }

    // prefetch next K-slab into cache while this one computes
    if (k0 + TK < K) {
      int pm = m0 + (tid & 127); if (pm >= M) pm = M - 1;
      __builtin_prefetch(&A[(long)pm * K + (k0 + TK)], 0, 1);
      int pk = k0 + TK + (tid >> 5); if (pk >= K) pk = K - 1;
      __builtin_prefetch(&W[(long)pk * Nout + n0], 0, 1);
    }
    __syncthreads();

    // ---- WMMA fragments: lane (g = lane>>4, r = lane&15)
    //      16-bit A/B layout: elems 0..7 <- K = g*8+0..7 ; elems 8..15 <- K = g*8+16..23
    const int lr = lane & 15;
    const int g8 = (lane >> 4) << 3;

    union Frag { bf16x16 v; bf16x8 h[2]; };
    Frag bf;
    bf.h[0] = *(const bf16x8*)&Bs[wn * 16 + lr][g8];
    bf.h[1] = *(const bf16x8*)&Bs[wn * 16 + lr][g8 + 16];

    #pragma unroll
    for (int s = 0; s < 4; ++s) {
      const int row = wm * 64 + s * 16 + lr;
      Frag af;
      af.h[0] = *(const bf16x8*)&As[row][g8];
      af.h[1] = *(const bf16x8*)&As[row][g8 + 16];
      acc[s].v = __builtin_amdgcn_wmma_f32_16x16x32_bf16(
          /*neg_a=*/false, af.v, /*neg_b=*/false, bf.v,
          /*c_mod=*/(short)0, acc[s].v, /*reuse_a=*/false, /*reuse_b=*/false);
    }
    __syncthreads();
  }

  // D layout: N = lane&15, M = vgpr + 8*(lane>>4)
  const int lcol = lane & 15;
  const int mo   = (lane >> 4) << 3;
  const int gn   = n0 + wn * 16 + lcol;
  if (gn < Nout) {
    const float bv = bias ? bias[gn] : 0.f;
    #pragma unroll
    for (int s = 0; s < 4; ++s)
      #pragma unroll
      for (int vi = 0; vi < 8; ++vi) {
        int gm = m0 + wm * 64 + s * 16 + vi + mo;
        if (gm < M) {
          float o = acc[s].f[vi] + bv;
          if (do_relu) o = fmaxf(o, 0.f);
          C[(long)gm * Nout + gn] = o;
        }
      }
  }
}

// ---------------------------------------------------------------------------
// Elementwise / fill helpers (grid-stride)
// ---------------------------------------------------------------------------
__global__ void fill_f32_kernel(float* p, float v, long n) {
  long i = (long)blockIdx.x * blockDim.x + threadIdx.x;
  long st = (long)gridDim.x * blockDim.x;
  for (; i < n; i += st) p[i] = v;
}
__global__ void fill_u32_kernel(unsigned* p, unsigned v, long n) {
  long i = (long)blockIdx.x * blockDim.x + threadIdx.x;
  long st = (long)gridDim.x * blockDim.x;
  for (; i < n; i += st) p[i] = v;
}

// h0[n, f*64+j] = emb[x[n,f], j]
__global__ void embed_kernel(const int* __restrict__ x, const float* __restrict__ emb,
                             float* __restrict__ h0, long n) {
  long i = (long)blockIdx.x * blockDim.x + threadIdx.x;
  long st = (long)gridDim.x * blockDim.x;
  for (; i < n; i += st) {
    long node = i >> 9;
    int  c    = (int)(i & 511);
    int  f    = c >> 6, j = c & 63;
    int  id   = x[node * 8 + f];
    h0[i] = emb[(long)id * 64 + j];
  }
}

// out = relu(attn + skip)
__global__ void add_relu_kernel(const float* __restrict__ a, const float* __restrict__ b,
                                float* __restrict__ o, long n) {
  long i = (long)blockIdx.x * blockDim.x + threadIdx.x;
  long st = (long)gridDim.x * blockDim.x;
  for (; i < n; i += st) o[i] = fmaxf(a[i] + b[i], 0.f);
}

// ---------------------------------------------------------------------------
// Edge attention: one wave32 per edge, 8 channels (of DH=256) per lane.
// Edge feature projection eattr[e,0:6] @ We[6,256] is computed inline.
// ---------------------------------------------------------------------------
__global__ __launch_bounds__(256)
void edge_alpha_kernel(const float* __restrict__ q, const float* __restrict__ k,
                       const int* __restrict__ src, const int* __restrict__ dst,
                       const float* __restrict__ eattr, const float* __restrict__ We,
                       float* __restrict__ alpha, unsigned* __restrict__ mmax, int E) {
  int e = blockIdx.x * 8 + (threadIdx.x >> 5);
  int lane = threadIdx.x & 31;
  if (e >= E) return;
  int s = src[e], d = dst[e];
  float ea[6];
  #pragma unroll
  for (int i = 0; i < 6; ++i) ea[i] = eattr[(long)e * 6 + i];
  float partial = 0.f;
  #pragma unroll
  for (int j = 0; j < 8; ++j) {
    int c = j * 32 + lane;
    float ec = 0.f;
    #pragma unroll
    for (int i = 0; i < 6; ++i) ec += ea[i] * We[i * 256 + c];
    partial += q[(long)d * 256 + c] * (k[(long)s * 256 + c] + ec);
  }
  for (int off = 16; off; off >>= 1) partial += __shfl_xor(partial, off, 32);
  if (lane == 0) {
    float a = partial * 0.0625f;  // 1/sqrt(256)
    alpha[e] = a;
    atomicMax(&mmax[d], f2ord(a));
  }
}

__global__ void edge_exp_kernel(const float* __restrict__ alpha, const int* __restrict__ dst,
                                const unsigned* __restrict__ mmax, float* __restrict__ aexp,
                                float* __restrict__ denom, int E) {
  int e = blockIdx.x * blockDim.x + threadIdx.x;
  if (e >= E) return;
  int d = dst[e];
  float a = __expf(alpha[e] - ord2f(mmax[d]));
  aexp[e] = a;
  atomicAdd(&denom[d], a);
}

__global__ __launch_bounds__(256)
void edge_scatter_kernel(const float* __restrict__ v, const float* __restrict__ aexp,
                         const float* __restrict__ denom,
                         const int* __restrict__ src, const int* __restrict__ dst,
                         const float* __restrict__ eattr, const float* __restrict__ We,
                         float* __restrict__ attn, int E) {
  int e = blockIdx.x * 8 + (threadIdx.x >> 5);
  int lane = threadIdx.x & 31;
  if (e >= E) return;
  int s = src[e], d = dst[e];
  float w = aexp[e] / denom[d];
  float ea[6];
  #pragma unroll
  for (int i = 0; i < 6; ++i) ea[i] = eattr[(long)e * 6 + i];
  #pragma unroll
  for (int j = 0; j < 8; ++j) {
    int c = j * 32 + lane;
    float ec = 0.f;
    #pragma unroll
    for (int i = 0; i < 6; ++i) ec += ea[i] * We[i * 256 + c];
    atomicAdd(&attn[(long)d * 256 + c], w * (v[(long)s * 256 + c] + ec));
  }
}

// ---------------------------------------------------------------------------
// Pooling (mean + max per batch segment) via atomics
// ---------------------------------------------------------------------------
__global__ void pool_acc_kernel(const float* __restrict__ H, const int* __restrict__ batch,
                                float* __restrict__ psum, unsigned* __restrict__ pmax,
                                float* __restrict__ cnt, long n) {
  long i = (long)blockIdx.x * blockDim.x + threadIdx.x;
  long st = (long)gridDim.x * blockDim.x;
  for (; i < n; i += st) {
    long node = i >> 8;
    int  c    = (int)(i & 255);
    int  b    = batch[node];
    float h   = H[i];
    atomicAdd(&psum[(long)b * 256 + c], h);
    atomicMax(&pmax[(long)b * 256 + c], f2ord(h));
    if (c == 0) atomicAdd(&cnt[b], 1.f);
  }
}

// out[b*ldo + c] = mean; out[b*ldo + 256 + c] = max
__global__ void pool_fin_kernel(const float* __restrict__ psum, const unsigned* __restrict__ pmax,
                                const float* __restrict__ cnt, float* __restrict__ out,
                                int ldo, int B) {
  int i = blockIdx.x * blockDim.x + threadIdx.x;
  if (i >= B * 256) return;
  int b = i >> 8, c = i & 255;
  out[(long)b * ldo + c]       = psum[i] / cnt[b];
  out[(long)b * ldo + 256 + c] = ord2f(pmax[i]);
}

// ---------------------------------------------------------------------------
// Tiny gate / head kernels (M=64 rows)
// ---------------------------------------------------------------------------
__global__ void gate_sig_kernel(const float* __restrict__ gh, const float* __restrict__ gW2,
                                const float* __restrict__ gb2, float* __restrict__ g, int B) {
  int b = blockIdx.x * blockDim.x + threadIdx.x;
  if (b >= B) return;
  float acc = gb2[0];
  for (int i = 0; i < 128; ++i) acc += gh[(long)b * 128 + i] * gW2[i];
  g[b] = 1.f / (1.f + __expf(-acc));
}

__global__ void mix_kernel(const float* __restrict__ g, const float* __restrict__ gcat,
                           float* __restrict__ rr, int B) {
  int i = blockIdx.x * blockDim.x + threadIdx.x;
  if (i >= B * 512) return;
  int b = i >> 9, c = i & 511;
  float gg  = g[b];
  float und = gcat[(long)b * 1024 + c];
  float dir = gcat[(long)b * 1024 + 512 + c];
  rr[i] = gg * dir + (1.f - gg) * und;
}

__global__ void head_out_kernel(const float* __restrict__ t2, const float* __restrict__ hW3,
                                const float* __restrict__ hb3, float* __restrict__ out, int B) {
  int b = blockIdx.x * blockDim.x + threadIdx.x;
  if (b >= B) return;
  float acc = hb3[0];
  for (int i = 0; i < 128; ++i) acc += t2[(long)b * 128 + i] * hW3[i];
  out[b] = acc;
}

// ---------------------------------------------------------------------------
// Host-side orchestration
// ---------------------------------------------------------------------------
static inline void launch_gemm(const float* A, const float* W, const float* bias, float* C,
                               int M, int K, int Nout, bool relu, hipStream_t st) {
  dim3 g((Nout + TN - 1) / TN, (M + TM - 1) / TM);
  gemm_bias_kernel<<<g, 256, 0, st>>>(A, W, bias, C, M, K, Nout, relu ? 1 : 0);
}
static inline void launch_fill(float* p, float v, long n, hipStream_t st) {
  int blocks = (int)((n + 255) / 256); if (blocks > 2048) blocks = 2048;
  fill_f32_kernel<<<blocks, 256, 0, st>>>(p, v, n);
}
static inline void launch_fillu(unsigned* p, unsigned v, long n, hipStream_t st) {
  int blocks = (int)((n + 255) / 256); if (blocks > 2048) blocks = 2048;
  fill_u32_kernel<<<blocks, 256, 0, st>>>(p, v, n);
}

struct ConvP { const float *Wq,*bq,*Wk,*bk,*Wv,*bv,*We,*Ws,*bs; };

static void run_tconv(const float* h, int din, const int* ei, const float* eattr,
                      const ConvP& p, int N, int E,
                      float* q, float* k, float* v, float* sk, float* attn,
                      float* alpha, float* aexp, unsigned* mmax, float* denom,
                      float* out, hipStream_t st) {
  launch_gemm(h, p.Wq, p.bq, q,  N, din, 256, false, st);
  launch_gemm(h, p.Wk, p.bk, k,  N, din, 256, false, st);
  launch_gemm(h, p.Wv, p.bv, v,  N, din, 256, false, st);
  launch_gemm(h, p.Ws, p.bs, sk, N, din, 256, false, st);

  launch_fill(attn, 0.f, (long)N * 256, st);
  launch_fill(denom, 0.f, N, st);
  launch_fillu(mmax, 0u, N, st);

  const int* src = ei;       // edge_index[0]
  const int* dst = ei + E;   // edge_index[1]
  edge_alpha_kernel<<<(E + 7) / 8, 256, 0, st>>>(q, k, src, dst, eattr, p.We, alpha, mmax, E);
  edge_exp_kernel<<<(E + 255) / 256, 256, 0, st>>>(alpha, dst, mmax, aexp, denom, E);
  edge_scatter_kernel<<<(E + 7) / 8, 256, 0, st>>>(v, aexp, denom, src, dst, eattr, p.We, attn, E);

  long n = (long)N * 256;
  int blocks = (int)((n + 255) / 256); if (blocks > 2048) blocks = 2048;
  add_relu_kernel<<<blocks, 256, 0, st>>>(attn, sk, out, n);
}

static void run_pool(const float* H, const int* batch, int N, int B,
                     float* psum, unsigned* pmax, float* cnt,
                     float* out_base, hipStream_t st) {
  launch_fill(psum, 0.f, (long)B * 256, st);
  launch_fillu(pmax, 0u, (long)B * 256, st);
  launch_fill(cnt, 0.f, B, st);
  long n = (long)N * 256;
  int blocks = (int)((n + 255) / 256); if (blocks > 2048) blocks = 2048;
  pool_acc_kernel<<<blocks, 256, 0, st>>>(H, batch, psum, pmax, cnt, n);
  pool_fin_kernel<<<(B * 256 + 255) / 256, 256, 0, st>>>(psum, pmax, cnt, out_base, 1024, B);
}

extern "C" void kernel_launch(void* const* d_in, const int* in_sizes, int n_in,
                              void* d_out, int out_size, void* d_ws, size_t ws_size,
                              hipStream_t stream) {
  const int* x       = (const int*)  d_in[0];
  const int* ei_dir  = (const int*)  d_in[1];
  const float* ea_d  = (const float*)d_in[2];
  const int* ei_und  = (const int*)  d_in[3];
  const float* ea_u  = (const float*)d_in[4];
  const int* batch   = (const int*)  d_in[5];
  const float* emb   = (const float*)d_in[6];

  auto conv = [&](int base) {
    ConvP p;
    p.Wq = (const float*)d_in[base + 0]; p.bq = (const float*)d_in[base + 1];
    p.Wk = (const float*)d_in[base + 2]; p.bk = (const float*)d_in[base + 3];
    p.Wv = (const float*)d_in[base + 4]; p.bv = (const float*)d_in[base + 5];
    p.We = (const float*)d_in[base + 6];
    p.Ws = (const float*)d_in[base + 7]; p.bs = (const float*)d_in[base + 8];
    return p;
  };
  ConvP und1 = conv(7), und2 = conv(16), dir1 = conv(25), dir2 = conv(34);
  const float* gW1 = (const float*)d_in[43]; const float* gb1 = (const float*)d_in[44];
  const float* gW2 = (const float*)d_in[45]; const float* gb2 = (const float*)d_in[46];
  const float* hW1 = (const float*)d_in[47]; const float* hb1 = (const float*)d_in[48];
  const float* hW2 = (const float*)d_in[49]; const float* hb2 = (const float*)d_in[50];
  const float* hW3 = (const float*)d_in[51]; const float* hb3 = (const float*)d_in[52];

  const int N = in_sizes[5];        // batch vector length
  const int E = in_sizes[2] / 6;    // edge_attr rows
  const int B = out_size;           // 64

  // workspace layout (floats)
  float* ws = (float*)d_ws;
  long o = 0;
  float* h0    = ws + o; o += (long)N * 512;
  float* q     = ws + o; o += (long)N * 256;
  float* k     = ws + o; o += (long)N * 256;
  float* v     = ws + o; o += (long)N * 256;
  float* sk    = ws + o; o += (long)N * 256;
  float* attn  = ws + o; o += (long)N * 256;
  float* hbuf  = ws + o; o += (long)N * 256;
  float* Hund  = ws + o; o += (long)N * 256;
  float* Hdir  = ws + o; o += (long)N * 256;
  float* alpha = ws + o; o += E;
  float* aexp  = ws + o; o += E;
  unsigned* mmax = (unsigned*)(ws + o); o += N;
  float* denom = ws + o; o += N;
  float* psum  = ws + o; o += (long)B * 256;
  unsigned* pmax = (unsigned*)(ws + o); o += (long)B * 256;
  float* cnt   = ws + o; o += B;
  float* gcat  = ws + o; o += (long)B * 1024;
  float* gh    = ws + o; o += (long)B * 128;
  float* gbuf  = ws + o; o += B;
  float* rr    = ws + o; o += (long)B * 512;
  float* t1    = ws + o; o += (long)B * 256;
  float* t2    = ws + o; o += (long)B * 128;
  (void)ws_size; (void)n_in;

  // 1) embedding lookup -> h0 [N, 512]
  {
    long n = (long)N * 512;
    int blocks = (int)((n + 255) / 256); if (blocks > 2048) blocks = 2048;
    embed_kernel<<<blocks, 256, 0, stream>>>(x, emb, h0, n);
  }

  // 2) undirected branch
  run_tconv(h0,   512, ei_und, ea_u, und1, N, E, q, k, v, sk, attn, alpha, aexp, mmax, denom, hbuf, stream);
  run_tconv(hbuf, 256, ei_und, ea_u, und2, N, E, q, k, v, sk, attn, alpha, aexp, mmax, denom, Hund, stream);

  // 3) directed branch
  run_tconv(h0,   512, ei_dir, ea_d, dir1, N, E, q, k, v, sk, attn, alpha, aexp, mmax, denom, hbuf, stream);
  run_tconv(hbuf, 256, ei_dir, ea_d, dir2, N, E, q, k, v, sk, attn, alpha, aexp, mmax, denom, Hdir, stream);

  // 4) pooling -> gcat = [r_und | r_dir] per batch row (B x 1024)
  run_pool(Hund, batch, N, B, psum, pmax, cnt, gcat,       stream);
  run_pool(Hdir, batch, N, B, psum, pmax, cnt, gcat + 512, stream);

  // 5) gate
  launch_gemm(gcat, gW1, gb1, gh, B, 1024, 128, true, stream);
  gate_sig_kernel<<<(B + 63) / 64, 64, 0, stream>>>(gh, gW2, gb2, gbuf, B);
  mix_kernel<<<(B * 512 + 255) / 256, 256, 0, stream>>>(gbuf, gcat, rr, B);

  // 6) head
  launch_gemm(rr, hW1, hb1, t1, B, 512, 256, true, stream);
  launch_gemm(t1, hW2, hb2, t2, B, 256, 128, true, stream);
  head_out_kernel<<<(B + 63) / 64, 64, 0, stream>>>(t2, hW3, hb3, (float*)d_out, B);
}
